// CrossFramelGraphFormerBlock_76948634075406
// MI455X (gfx1250) — compile-verified, matrix operand order checked by reading, not run on hardware
//
#include <hip/hip_runtime.h>

// ---------------------------------------------------------------------------
// Types / WMMA helpers (gfx1250, wave32)
// ---------------------------------------------------------------------------
typedef _Float16 h16;
typedef __attribute__((ext_vector_type(16))) _Float16 v16h;
typedef __attribute__((ext_vector_type(8)))  float    v8f;

#define LDSS 40  // LDS tile row stride in halves (80B -> rows 16B aligned)

union FragAB { v16h v; unsigned int u[8]; };

__device__ __forceinline__ int imin_i(int a, int b) { return a < b ? a : b; }

// Build a 16x32 f16 fragment from an LDS tile (row-major, stride LDSS halves).
// Layout per CDNA5 ISA 7.12.2: lane = M (or N), VGPR v<4 -> K = 2v + 8*hi,
// VGPR v>=4 -> K = 16 + 2(v-4) + 8*hi, two halves per VGPR.
__device__ __forceinline__ void load_frag(const h16* lds, int row, FragAB& f) {
  const int lane = threadIdx.x & 31;
  const int m = lane & 15;
  const int hi = lane >> 4;
  const h16* p = lds + (row + m) * LDSS + 8 * hi;
#pragma unroll
  for (int v = 0; v < 4; ++v) f.u[v] = *(const unsigned int*)(p + 2 * v);
#pragma unroll
  for (int v = 0; v < 4; ++v) f.u[4 + v] = *(const unsigned int*)(p + 16 + 2 * v);
}

__device__ __forceinline__ v8f wmma16(const FragAB& a, const FragAB& b, v8f c) {
  return __builtin_amdgcn_wmma_f32_16x16x32_f16(false, a.v, false, b.v,
                                                (short)0, c, false, false);
}

__device__ __forceinline__ float qgelu_f(float v) {
  return v / (1.0f + __expf(-1.702f * v));
}

// ---------------------------------------------------------------------------
// GEMM 1: C[M,N] = act(A_fp32[M,K] * B_f16[N,K]^T + bias) (+resid), store f32/f16
// Block: 128 threads (4 waves), 64x64 tile, K-step 32.
// Staging is branchless and phase-split: all global loads of a k-slab issue
// first (kept in flight together), then convert + LDS store.
// ---------------------------------------------------------------------------
__global__ __launch_bounds__(128)
void gemm_xw_kernel(const float* __restrict__ A, int lda,
                    const h16* __restrict__ B, int ldb,
                    const float* __restrict__ bias,
                    const float* __restrict__ resid, int ldr,
                    float* __restrict__ Cf, h16* __restrict__ Ch, int ldc,
                    int M, int N, int K, int act) {
  __shared__ h16 lA[64 * LDSS];
  __shared__ h16 lB[64 * LDSS];
  const int tid = threadIdx.x;
  const int lane = tid & 31;
  const int wave = tid >> 5;
  const int wm = (wave >> 1) * 32;
  const int wn = (wave & 1) * 32;
  const int mBase = blockIdx.y * 64;
  const int nBase = blockIdx.x * 64;

  const int colA = (tid & 7) * 4;
  const int colB = (tid & 3) * 8;

  v8f acc[2][2] = {};

  for (int kb = 0; kb < K; kb += 32) {
    // Phase 1: issue all global loads of this k-slab
    float4 fa[4];
    bool vA[4];
#pragma unroll
    for (int r = 0; r < 4; ++r) {
      const int row = r * 16 + (tid >> 3);
      const int gr = mBase + row;
      const int cr = imin_i(gr, M - 1);
      const float* ap = A + (size_t)cr * lda + kb + colA;
      fa[r] = *(const float4*)ap;
      __builtin_prefetch(ap + 32, 0, 3);
      vA[r] = (gr < M);
    }
    uint4 ub[2];
    bool vB[2];
#pragma unroll
    for (int r = 0; r < 2; ++r) {
      const int row = r * 32 + (tid >> 2);
      const int gr = nBase + row;
      const int cr = imin_i(gr, N - 1);
      ub[r] = *(const uint4*)(B + (size_t)cr * ldb + kb + colB);
      vB[r] = (gr < N);
    }
    // Phase 2: convert + store to LDS
#pragma unroll
    for (int r = 0; r < 4; ++r) {
      const int row = r * 16 + (tid >> 3);
      h16* lp = &lA[row * LDSS + colA];
      lp[0] = (h16)(vA[r] ? fa[r].x : 0.0f);
      lp[1] = (h16)(vA[r] ? fa[r].y : 0.0f);
      lp[2] = (h16)(vA[r] ? fa[r].z : 0.0f);
      lp[3] = (h16)(vA[r] ? fa[r].w : 0.0f);
    }
#pragma unroll
    for (int r = 0; r < 2; ++r) {
      const int row = r * 32 + (tid >> 2);
      uint4 u = ub[r];
      if (!vB[r]) { u.x = 0u; u.y = 0u; u.z = 0u; u.w = 0u; }
      *(uint4*)&lB[row * LDSS + colB] = u;
    }
    __syncthreads();
    FragAB a0, a1, b0, b1;
    load_frag(lA, wm, a0);
    load_frag(lA, wm + 16, a1);
    load_frag(lB, wn, b0);
    load_frag(lB, wn + 16, b1);
    acc[0][0] = wmma16(a0, b0, acc[0][0]);
    acc[0][1] = wmma16(a0, b1, acc[0][1]);
    acc[1][0] = wmma16(a1, b0, acc[1][0]);
    acc[1][1] = wmma16(a1, b1, acc[1][1]);
    __syncthreads();
  }

  const int hi = lane >> 4;
  const int cn = lane & 15;
#pragma unroll
  for (int i = 0; i < 2; ++i)
#pragma unroll
    for (int j = 0; j < 2; ++j)
#pragma unroll
      for (int r = 0; r < 8; ++r) {
        const int grow = mBase + wm + 16 * i + r + 8 * hi;
        const int gcol = nBase + wn + 16 * j + cn;
        if (grow < M && gcol < N) {
          float v = acc[i][j][r];
          if (bias) v += bias[gcol];
          if (act == 1) v = qgelu_f(v);
          if (resid) v += resid[(size_t)grow * ldr + gcol];
          if (Cf) Cf[(size_t)grow * ldc + gcol] = v;
          if (Ch) Ch[(size_t)grow * ldc + gcol] = (h16)v;
        }
      }
}

// ---------------------------------------------------------------------------
// GEMM 2 (batched attention GEMM): C = alpha * A_f16 * op(B_f16)
//   z -> (bt = z/HH, h = z%HH); pointer offsets via per-bt / per-h strides.
//   transB==0: op(B)=B^T with B given as [N,K] rows (QK^T, sim)
//   transB==1: op(B)=B    with B given as [K,N] rows (P*V)
// Operands live in the (slack-padded) workspace -> row-clamped wide loads with
// select-to-zero k-tails are safe. Loads phase-split like gemm_xw.
// ---------------------------------------------------------------------------
__global__ __launch_bounds__(128)
void gemm_f16_kernel(const h16* __restrict__ A, int lda, long long sBtA, long long sHA,
                     const h16* __restrict__ B, int ldb, long long sBtB, long long sHB,
                     float* __restrict__ C, int ldc, long long sBtC, long long sHC,
                     int M, int N, int K, int HH, float alpha, int transB) {
  __shared__ h16 lA[64 * LDSS];
  __shared__ h16 lB[64 * LDSS];
  const int z = blockIdx.z;
  const int bt = z / HH;
  const int hid = z % HH;
  A += (size_t)bt * sBtA + (size_t)hid * sHA;
  B += (size_t)bt * sBtB + (size_t)hid * sHB;
  C += (size_t)bt * sBtC + (size_t)hid * sHC;

  const int tid = threadIdx.x;
  const int lane = tid & 31;
  const int wave = tid >> 5;
  const int wm = (wave >> 1) * 32;
  const int wn = (wave & 1) * 32;
  const int mBase = blockIdx.y * 64;
  const int nBase = blockIdx.x * 64;

  const int colA = (tid & 3) * 8;

  v8f acc[2][2] = {};

  for (int kb = 0; kb < K; kb += 32) {
    // Phase 1: issue all global loads of this k-slab
    h16 ta[2][8] __attribute__((aligned(16)));
    bool vA[2];
#pragma unroll
    for (int r = 0; r < 2; ++r) {
      const int row = r * 32 + (tid >> 2);
      const int gr = mBase + row;
      const int cr = imin_i(gr, M - 1);
      const h16* ap = A + (size_t)cr * lda + kb + colA;
      *(uint4*)ta[r] = *(const uint4*)ap;
      __builtin_prefetch(ap + 32, 0, 3);
      vA[r] = (gr < M);
    }
    h16 tb[2][8] __attribute__((aligned(16)));
    bool vB[2];
    if (!transB) {
#pragma unroll
      for (int r = 0; r < 2; ++r) {
        const int row = r * 32 + (tid >> 2);
        const int gr = nBase + row;
        const int cr = imin_i(gr, N - 1);
        *(uint4*)tb[r] = *(const uint4*)(B + (size_t)cr * ldb + kb + colA);
        vB[r] = (gr < N);
      }
    } else {
#pragma unroll
      for (int r = 0; r < 2; ++r) {
        const int kk = r * 16 + (tid >> 3);
        const int gk = kb + kk;
        const int ck = imin_i(gk, K - 1);
        *(uint4*)tb[r] = *(const uint4*)(B + (size_t)ck * ldb + nBase + (tid & 7) * 8);
        vB[r] = (gk < K);
      }
    }
    // Phase 2: store to LDS (select-to-zero for OOB rows / k-tail)
#pragma unroll
    for (int r = 0; r < 2; ++r) {
      const int row = r * 32 + (tid >> 2);
      h16* lp = &lA[row * LDSS + colA];
#pragma unroll
      for (int q = 0; q < 8; ++q)
        lp[q] = (vA[r] && (kb + colA + q) < K) ? ta[r][q] : (h16)0.0f;
    }
    if (!transB) {
#pragma unroll
      for (int r = 0; r < 2; ++r) {
        const int row = r * 32 + (tid >> 2);
        h16* lp = &lB[row * LDSS + colA];
#pragma unroll
        for (int q = 0; q < 8; ++q)
          lp[q] = (vB[r] && (kb + colA + q) < K) ? tb[r][q] : (h16)0.0f;
      }
    } else {
#pragma unroll
      for (int r = 0; r < 2; ++r) {
        const int kk = r * 16 + (tid >> 3);
        const int nn = (tid & 7) * 8;
#pragma unroll
        for (int q = 0; q < 8; ++q)
          lB[(nn + q) * LDSS + kk] = vB[r] ? tb[r][q] : (h16)0.0f;
      }
    }
    __syncthreads();
    FragAB a0, a1, b0, b1;
    load_frag(lA, wm, a0);
    load_frag(lA, wm + 16, a1);
    load_frag(lB, wn, b0);
    load_frag(lB, wn + 16, b1);
    acc[0][0] = wmma16(a0, b0, acc[0][0]);
    acc[0][1] = wmma16(a0, b1, acc[0][1]);
    acc[1][0] = wmma16(a1, b0, acc[1][0]);
    acc[1][1] = wmma16(a1, b1, acc[1][1]);
    __syncthreads();
  }

  const int hi = lane >> 4;
  const int cn = lane & 15;
#pragma unroll
  for (int i = 0; i < 2; ++i)
#pragma unroll
    for (int j = 0; j < 2; ++j)
#pragma unroll
      for (int r = 0; r < 8; ++r) {
        const int grow = mBase + wm + 16 * i + r + 8 * hi;
        const int gcol = nBase + wn + 16 * j + cn;
        if (grow < M && gcol < N)
          C[(size_t)grow * ldc + gcol] = alpha * acc[i][j][r];
      }
}

// ---------------------------------------------------------------------------
// Softmax kernels (row per block.x, batch per block.y), probs written as f16
// ---------------------------------------------------------------------------
__global__ __launch_bounds__(256)
void softmax_plain_kernel(const float* __restrict__ S, int ldS, long long sZ,
                          h16* __restrict__ P, int ldP, long long pZ, int cols) {
  const int row = blockIdx.x;
  const int z = blockIdx.y;
  const float* s = S + (size_t)z * sZ + (size_t)row * ldS;
  h16* p = P + (size_t)z * pZ + (size_t)row * ldP;
  __shared__ float red[256];
  const int tid = threadIdx.x;
  float loc[8];
  const int nIt = (cols + 255) >> 8;
  float mx = -3.0e38f;
  for (int i = 0; i < nIt; ++i) {
    const int c = i * 256 + tid;
    const float v = (c < cols) ? s[c] : -3.0e38f;
    loc[i] = v;
    mx = fmaxf(mx, v);
  }
  red[tid] = mx; __syncthreads();
  for (int o = 128; o > 0; o >>= 1) {
    if (tid < o) red[tid] = fmaxf(red[tid], red[tid + o]);
    __syncthreads();
  }
  mx = red[0]; __syncthreads();
  float sum = 0.0f;
  for (int i = 0; i < nIt; ++i) {
    const int c = i * 256 + tid;
    const float e = (c < cols) ? __expf(loc[i] - mx) : 0.0f;
    loc[i] = e;
    sum += e;
  }
  red[tid] = sum; __syncthreads();
  for (int o = 128; o > 0; o >>= 1) {
    if (tid < o) red[tid] += red[tid + o];
    __syncthreads();
  }
  const float inv = 1.0f / red[0];
  for (int i = 0; i < nIt; ++i) {
    const int c = i * 256 + tid;
    if (c < cols) p[c] = (h16)(loc[i] * inv);
  }
}

// Graph softmax: adj = relu(sim)*mask computed inline; score -> -inf where adj==0
__global__ __launch_bounds__(256)
void softmax_graph_kernel(const float* __restrict__ S, int ldS, long long sZ,
                          const float* __restrict__ sim, int ldsim,
                          const float* __restrict__ mask, int ldm,
                          h16* __restrict__ P, int ldP, long long pZ, int cols) {
  const int row = blockIdx.x;
  const int z = blockIdx.y;
  const float* s = S + (size_t)z * sZ + (size_t)row * ldS;
  const float* sr = sim + (size_t)row * ldsim;
  const float* mr = mask + (size_t)row * ldm;
  h16* p = P + (size_t)z * pZ + (size_t)row * ldP;
  __shared__ float red[256];
  const int tid = threadIdx.x;
  float loc[8];
  const int nIt = (cols + 255) >> 8;
  float mx = -3.0e38f;
  for (int i = 0; i < nIt; ++i) {
    const int c = i * 256 + tid;
    float v = -3.0e38f;
    if (c < cols) {
      const float sv = sr[c];
      const float adj = (sv < 0.0f ? 0.0f : sv) * mr[c];
      if (adj != 0.0f) v = s[c] * adj;
    }
    loc[i] = v;
    mx = fmaxf(mx, v);
  }
  red[tid] = mx; __syncthreads();
  for (int o = 128; o > 0; o >>= 1) {
    if (tid < o) red[tid] = fmaxf(red[tid], red[tid + o]);
    __syncthreads();
  }
  mx = red[0]; __syncthreads();
  float sum = 0.0f;
  for (int i = 0; i < nIt; ++i) {
    const int c = i * 256 + tid;
    const float e = (c < cols && loc[i] > -1.0e38f) ? __expf(loc[i] - mx) : 0.0f;
    loc[i] = e;
    sum += e;
  }
  red[tid] = sum; __syncthreads();
  for (int o = 128; o > 0; o >>= 1) {
    if (tid < o) red[tid] += red[tid + o];
    __syncthreads();
  }
  const float inv = 1.0f / red[0];
  for (int i = 0; i < nIt; ++i) {
    const int c = i * 256 + tid;
    if (c < cols) p[c] = (h16)(loc[i] * inv);
  }
}

// ---------------------------------------------------------------------------
// LayerNorm / L2-normalize: one wave32 per 512-wide row
// ---------------------------------------------------------------------------
__global__ __launch_bounds__(256)
void layernorm_kernel(const float* __restrict__ X, const float* __restrict__ g,
                      const float* __restrict__ bta, float* __restrict__ Y, int rows) {
  const int row = blockIdx.x * 8 + (threadIdx.x >> 5);
  const int lane = threadIdx.x & 31;
  if (row >= rows) return;
  const float* x = X + (size_t)row * 512;
  float s = 0.0f, ss = 0.0f;
  for (int i = lane; i < 512; i += 32) {
    const float v = x[i];
    s += v; ss += v * v;
  }
  for (int o = 16; o > 0; o >>= 1) {
    s += __shfl_xor(s, o, 32);
    ss += __shfl_xor(ss, o, 32);
  }
  const float m = s * (1.0f / 512.0f);
  const float var = ss * (1.0f / 512.0f) - m * m;
  const float inv = rsqrtf(var + 1e-5f);
  float* y = Y + (size_t)row * 512;
  for (int i = lane; i < 512; i += 32) y[i] = (x[i] - m) * inv * g[i] + bta[i];
}

__global__ __launch_bounds__(256)
void rownorm_kernel(const float* __restrict__ X, h16* __restrict__ Y, int rows) {
  const int row = blockIdx.x * 8 + (threadIdx.x >> 5);
  const int lane = threadIdx.x & 31;
  if (row >= rows) return;
  const float* x = X + (size_t)row * 512;
  float ss = 0.0f;
  for (int i = lane; i < 512; i += 32) {
    const float v = x[i];
    ss += v * v;
  }
  for (int o = 16; o > 0; o >>= 1) ss += __shfl_xor(ss, o, 32);
  const float inv = 1.0f / fmaxf(sqrtf(ss), 1e-12f);
  h16* y = Y + (size_t)row * 512;
  for (int i = lane; i < 512; i += 32) y[i] = (h16)(x[i] * inv);
}

// ---------------------------------------------------------------------------
// Tiny scalar MHA for the 32-token "message" path (S=8, Bt=4, H=8, DK=64)
// qkv rows indexed s*4+bt, cols [q|k|v] of 512 each. One block per (bt,h).
// ---------------------------------------------------------------------------
__global__ __launch_bounds__(64)
void glob_attn_kernel(const float* __restrict__ qkv, float* __restrict__ O) {
  const int z = blockIdx.x;
  const int bt = z & 3;
  const int h = z >> 2;
  __shared__ float sc[8][8];
  __shared__ float pr[8][8];
  const int s = threadIdx.x >> 3;
  const int t = threadIdx.x & 7;
  const float* q = qkv + (size_t)(s * 4 + bt) * 1536 + h * 64;
  const float* k = qkv + (size_t)(t * 4 + bt) * 1536 + 512 + h * 64;
  float d = 0.0f;
  for (int i = 0; i < 64; ++i) d += q[i] * k[i];
  sc[s][t] = d * 0.125f;
  __syncthreads();
  if (t == 0) {
    float mx = -3.0e38f;
    for (int j = 0; j < 8; ++j) mx = fmaxf(mx, sc[s][j]);
    float sum = 0.0f;
    for (int j = 0; j < 8; ++j) {
      const float e = __expf(sc[s][j] - mx);
      pr[s][j] = e;
      sum += e;
    }
    const float inv = 1.0f / sum;
    for (int j = 0; j < 8; ++j) pr[s][j] *= inv;
  }
  __syncthreads();
  const int dbase = t * 8;
  for (int qd = 0; qd < 8; ++qd) {
    const int dd = dbase + qd;
    float o = 0.0f;
    for (int j = 0; j < 8; ++j)
      o += pr[s][j] * qkv[(size_t)(j * 4 + bt) * 1536 + 1024 + h * 64 + dd];
    O[(size_t)(s * 4 + bt) * 512 + h * 64 + dd] = o;
  }
}

// ---------------------------------------------------------------------------
// Elementwise / layout kernels
// ---------------------------------------------------------------------------
__global__ void f32_to_f16_kernel(const float* __restrict__ in, h16* __restrict__ out, int n) {
  const int i = blockIdx.x * 256 + threadIdx.x;
  if (i < n) out[i] = (h16)in[i];
}

// glob[t*4+b] = xp[(b*8+t)*197 + 0]
__global__ void gather_glob_kernel(const float* __restrict__ xp, float* __restrict__ glob) {
  const int r = blockIdx.x;
  const int t = r >> 2, b = r & 3;
  const size_t src = (size_t)((b * 8 + t) * 197) * 512;
  for (int i = threadIdx.x; i < 512; i += 256) glob[(size_t)r * 512 + i] = xp[src + i];
}

// loc[b*1568 + t*196 + j] = xp[(b*8+t)*197 + 1 + j]
__global__ void gather_loc_kernel(const float* __restrict__ xp, float* __restrict__ loc) {
  const int r = blockIdx.x;
  const int b = r / 1568;
  const int rem = r % 1568;
  const int t = rem / 196;
  const int j = rem % 196;
  const size_t src = (size_t)((b * 8 + t) * 197 + 1 + j) * 512;
  for (int i = threadIdx.x; i < 512; i += 256) loc[(size_t)r * 512 + i] = xp[src + i];
}

// y[l*32 + t*4 + b] = (l==0) ? glob[t*4+b] : loc_o[b*1568 + t*196 + l-1]
__global__ void build_y_kernel(const float* __restrict__ glob, const float* __restrict__ loc_o,
                               float* __restrict__ y) {
  const int r = blockIdx.x;
  const int l = r >> 5;
  const int rem = r & 31;
  const int t = rem >> 2, b = rem & 3;
  const float* src = (l == 0)
                         ? (glob + (size_t)rem * 512)
                         : (loc_o + (size_t)(b * 1568 + t * 196 + (l - 1)) * 512);
  for (int i = threadIdx.x; i < 512; i += 256) y[(size_t)r * 512 + i] = src[i];
}

// out[(b*8+t)*197 + l] = y[l*32 + t*4 + b]
__global__ void scatter_out_kernel(const float* __restrict__ y, float* __restrict__ out) {
  const int r = blockIdx.x;
  const int bt8 = r / 197;
  const int l = r % 197;
  const int b = bt8 >> 3, t = bt8 & 7;
  const float* src = y + (size_t)(l * 32 + t * 4 + b) * 512;
  for (int i = threadIdx.x; i < 512; i += 256) out[(size_t)r * 512 + i] = src[i];
}

// ---------------------------------------------------------------------------
// Host orchestration
// ---------------------------------------------------------------------------
extern "C" void kernel_launch(void* const* d_in, const int* in_sizes, int n_in,
                              void* d_out, int out_size, void* d_ws, size_t ws_size,
                              hipStream_t stream) {
  (void)in_sizes; (void)n_in; (void)out_size; (void)ws_size;

  const float* x      = (const float*)d_in[0];
  const float* proj_w = (const float*)d_in[1];
  const float* proj_b = (const float*)d_in[2];
  const float* m_ln1g = (const float*)d_in[3];
  const float* m_ln1b = (const float*)d_in[4];
  const float* m_inw  = (const float*)d_in[5];
  const float* m_inb  = (const float*)d_in[6];
  const float* m_outw = (const float*)d_in[7];
  const float* m_outb = (const float*)d_in[8];
  const float* m_ln2g = (const float*)d_in[9];
  const float* m_ln2b = (const float*)d_in[10];
  const float* m_fcw  = (const float*)d_in[11];
  const float* m_fcb  = (const float*)d_in[12];
  const float* m_pw   = (const float*)d_in[13];
  const float* m_pb   = (const float*)d_in[14];
  const float* g_wq   = (const float*)d_in[15];
  const float* g_bq   = (const float*)d_in[16];
  const float* g_wk   = (const float*)d_in[17];
  const float* g_bk   = (const float*)d_in[18];
  const float* g_wv   = (const float*)d_in[19];
  const float* g_bv   = (const float*)d_in[20];
  const float* g_wo   = (const float*)d_in[21];
  const float* g_bo   = (const float*)d_in[22];
  const float* ln1g   = (const float*)d_in[23];
  const float* ln1b   = (const float*)d_in[24];
  const float* a_inw  = (const float*)d_in[25];
  const float* a_inb  = (const float*)d_in[26];
  const float* a_outw = (const float*)d_in[27];
  const float* a_outb = (const float*)d_in[28];
  const float* ln2g   = (const float*)d_in[29];
  const float* ln2b   = (const float*)d_in[30];
  const float* fcw    = (const float*)d_in[31];
  const float* fcb    = (const float*)d_in[32];
  const float* pw     = (const float*)d_in[33];
  const float* pb     = (const float*)d_in[34];
  const float* g_mask = (const float*)d_in[35];
  float* out = (float*)d_out;

  const int Ntok = 1568;            // local tokens per batch
  const int TOK  = 4 * 8 * 197;     // 6304 total tokens
  const int LOC  = 4 * Ntok;        // 6272
  const int YR   = 197 * 32;        // 6304 rows of y

  char* wsb = (char*)d_ws;
  size_t woff = 0;
  // Bump allocator with 512B slack per buffer so clamped k-tail vector loads
  // in the f16 GEMM never leave mapped memory.
  auto alloc = [&](size_t bytes) -> void* {
    void* p = wsb + woff;
    woff = (woff + bytes + 512 + 255) & ~(size_t)255;
    return p;
  };

  // f16 weights
  h16* wh_proj = (h16*)alloc((size_t)512 * 512 * 2);
  h16* wh_min  = (h16*)alloc((size_t)1536 * 512 * 2);
  h16* wh_mout = (h16*)alloc((size_t)512 * 512 * 2);
  h16* wh_mfc  = (h16*)alloc((size_t)2048 * 512 * 2);
  h16* wh_mpw  = (h16*)alloc((size_t)512 * 2048 * 2);
  h16* wh_gq   = (h16*)alloc((size_t)512 * 512 * 2);
  h16* wh_gk   = (h16*)alloc((size_t)512 * 512 * 2);
  h16* wh_gv   = (h16*)alloc((size_t)512 * 512 * 2);
  h16* wh_go   = (h16*)alloc((size_t)512 * 512 * 2);
  h16* wh_ain  = (h16*)alloc((size_t)1536 * 512 * 2);
  h16* wh_aout = (h16*)alloc((size_t)512 * 512 * 2);
  h16* wh_fc   = (h16*)alloc((size_t)2048 * 512 * 2);
  h16* wh_pw   = (h16*)alloc((size_t)512 * 2048 * 2);

  // activations
  float* xp    = (float*)alloc((size_t)TOK * 512 * 4);
  float* g0    = (float*)alloc((size_t)32 * 512 * 4);
  float* g1    = (float*)alloc((size_t)32 * 512 * 4);
  float* g2    = (float*)alloc((size_t)32 * 512 * 4);
  float* gt    = (float*)alloc((size_t)32 * 512 * 4);
  float* qkvg  = (float*)alloc((size_t)32 * 1536 * 4);
  float* aO    = (float*)alloc((size_t)32 * 512 * 4);
  float* hh32  = (float*)alloc((size_t)32 * 2048 * 4);
  float* loc   = (float*)alloc((size_t)LOC * 512 * 4);
  h16*   xnh   = (h16*)alloc((size_t)LOC * 512 * 2);
  float* sim   = (float*)alloc((size_t)4 * Ntok * Ntok * 4);
  h16*   qg    = (h16*)alloc((size_t)LOC * 512 * 2);
  h16*   kg    = (h16*)alloc((size_t)LOC * 512 * 2);
  h16*   vg    = (h16*)alloc((size_t)LOC * 512 * 2);
  float* gsc   = (float*)alloc((size_t)8 * Ntok * Ntok * 4);
  h16*   gpr   = (h16*)alloc((size_t)8 * Ntok * Ntok * 2);
  float* gO    = (float*)alloc((size_t)LOC * 512 * 4);
  float* loc_o = (float*)alloc((size_t)LOC * 512 * 4);
  float* yb    = (float*)alloc((size_t)YR * 512 * 4);
  float* ylb   = (float*)alloc((size_t)YR * 512 * 4);
  h16*   qkvh  = (h16*)alloc((size_t)YR * 1536 * 2);
  float* fsc   = (float*)alloc((size_t)256 * 197 * 208 * 4);
  h16*   fpr   = (h16*)alloc((size_t)256 * 197 * 208 * 2);
  float* fO    = (float*)alloc((size_t)YR * 512 * 4);
  float* y2    = (float*)alloc((size_t)YR * 512 * 4);
  float* hb    = (float*)alloc((size_t)YR * 2048 * 4);
  float* y3    = (float*)alloc((size_t)YR * 512 * 4);

  auto conv = [&](const float* src, h16* dst, int n) {
    f32_to_f16_kernel<<<dim3((n + 255) / 256), dim3(256), 0, stream>>>(src, dst, n);
  };
  auto ln = [&](const float* X, const float* g, const float* b, float* Y, int rows) {
    layernorm_kernel<<<dim3((rows + 7) / 8), dim3(256), 0, stream>>>(X, g, b, Y, rows);
  };
  auto gemm = [&](const float* A, int lda, const h16* Bm, int ldb, const float* bias,
                  const float* resid, int ldr, float* Cf, h16* Ch, int ldc,
                  int M, int N, int K, int act) {
    dim3 grid((N + 63) / 64, (M + 63) / 64, 1);
    gemm_xw_kernel<<<grid, dim3(128), 0, stream>>>(A, lda, Bm, ldb, bias, resid, ldr,
                                                   Cf, Ch, ldc, M, N, K, act);
  };
  auto gemm16 = [&](const h16* A, int lda, long long sBtA, long long sHA,
                    const h16* Bm, int ldb, long long sBtB, long long sHB,
                    float* C, int ldc, long long sBtC, long long sHC,
                    int M, int N, int K, int Hh, int Z, float alpha, int transB) {
    dim3 grid((N + 63) / 64, (M + 63) / 64, Z);
    gemm_f16_kernel<<<grid, dim3(128), 0, stream>>>(A, lda, sBtA, sHA, Bm, ldb, sBtB, sHB,
                                                    C, ldc, sBtC, sHC, M, N, K, Hh, alpha, transB);
  };

  // --- 0. weights -> f16 ---
  conv(proj_w, wh_proj, 512 * 512);
  conv(m_inw,  wh_min,  1536 * 512);
  conv(m_outw, wh_mout, 512 * 512);
  conv(m_fcw,  wh_mfc,  2048 * 512);
  conv(m_pw,   wh_mpw,  512 * 2048);
  conv(g_wq,   wh_gq,   512 * 512);
  conv(g_wk,   wh_gk,   512 * 512);
  conv(g_wv,   wh_gv,   512 * 512);
  conv(g_wo,   wh_go,   512 * 512);
  conv(a_inw,  wh_ain,  1536 * 512);
  conv(a_outw, wh_aout, 512 * 512);
  conv(fcw,    wh_fc,   2048 * 512);
  conv(pw,     wh_pw,   512 * 2048);

  // --- 1. token projection ---
  gemm(x, 512, wh_proj, 512, proj_b, nullptr, 0, xp, nullptr, 512, TOK, 512, 512, 0);

  // --- 2. split glob / loc ---
  gather_glob_kernel<<<32, 256, 0, stream>>>(xp, g0);
  gather_loc_kernel<<<LOC, 256, 0, stream>>>(xp, loc);

  // --- 3. message resblocks x2 (shared weights) ---
  const float* msrc = g0;
  float* mA = g1;
  float* mB = g2;
  for (int it = 0; it < 2; ++it) {
    ln(msrc, m_ln1g, m_ln1b, gt, 32);
    gemm(gt, 512, wh_min, 512, m_inb, nullptr, 0, qkvg, nullptr, 1536, 32, 1536, 512, 0);
    glob_attn_kernel<<<32, 64, 0, stream>>>(qkvg, aO);
    gemm(aO, 512, wh_mout, 512, m_outb, msrc, 512, mA, nullptr, 512, 32, 512, 512, 0);
    ln(mA, m_ln2g, m_ln2b, gt, 32);
    gemm(gt, 512, wh_mfc, 512, m_fcb, nullptr, 0, hh32, nullptr, 2048, 32, 2048, 512, 1);
    gemm(hh32, 2048, wh_mpw, 2048, m_pb, mA, 512, mB, nullptr, 512, 32, 512, 2048, 0);
    msrc = mB;
    mB = (it == 0) ? g0 : g2;  // ping-pong; g0 free after first iteration
  }
  const float* globF = msrc;

  // --- 4. graph build: xn = loc / max(||loc||,eps); sim = xn xn^T (batched z=4) ---
  rownorm_kernel<<<dim3((LOC + 7) / 8), 256, 0, stream>>>(loc, xnh, LOC);
  gemm16(xnh, 512, (long long)Ntok * 512, 0, xnh, 512, (long long)Ntok * 512, 0,
         sim, Ntok, (long long)Ntok * Ntok, 0, Ntok, Ntok, 512, 1, 4, 1.0f, 0);

  // --- 5. graph QKV projections (f16 outputs for attention) ---
  gemm(loc, 512, wh_gq, 512, g_bq, nullptr, 0, nullptr, qg, 512, LOC, 512, 512, 0);
  gemm(loc, 512, wh_gk, 512, g_bk, nullptr, 0, nullptr, kg, 512, LOC, 512, 512, 0);
  gemm(loc, 512, wh_gv, 512, g_bv, nullptr, 0, nullptr, vg, 512, LOC, 512, 512, 0);

  // --- 6. graph attention, per batch (z = 8 heads), scratch reused across b ---
  for (int b = 0; b < 4; ++b) {
    const h16* qb = qg + (size_t)b * Ntok * 512;
    const h16* kb = kg + (size_t)b * Ntok * 512;
    const h16* vb = vg + (size_t)b * Ntok * 512;
    gemm16(qb, 512, 0, 64, kb, 512, 0, 64,
           gsc, Ntok, 0, (long long)Ntok * Ntok, Ntok, Ntok, 64, 8, 8, 0.125f, 0);
    softmax_graph_kernel<<<dim3(Ntok, 8), 256, 0, stream>>>(
        gsc, Ntok, (long long)Ntok * Ntok,
        sim + (size_t)b * Ntok * Ntok, Ntok, g_mask, Ntok,
        gpr, Ntok, (long long)Ntok * Ntok, Ntok);
    gemm16(gpr, Ntok, 0, (long long)Ntok * Ntok, vb, 512, 0, 64,
           gO + (size_t)b * Ntok * 512, 512, 0, 64, Ntok, 64, Ntok, 8, 8, 1.0f, 1);
  }

  // --- 7. graph output projection (no residual) ---
  gemm(gO, 512, wh_go, 512, g_bo, nullptr, 0, loc_o, nullptr, 512, LOC, 512, 512, 0);

  // --- 8. assemble y = concat(glob, loc) in (L, T*B, D) order ---
  build_y_kernel<<<YR, 256, 0, stream>>>(globF, loc_o, yb);

  // --- 9. final transformer block: attention (S=197, batch 32, H=8) ---
  ln(yb, ln1g, ln1b, ylb, YR);
  gemm(ylb, 512, wh_ain, 512, a_inb, nullptr, 0, nullptr, qkvh, 1536, YR, 1536, 512, 0);
  gemm16(qkvh, 32 * 1536, 1536, 64, qkvh + 512, 32 * 1536, 1536, 64,
         fsc, 208, (long long)8 * 197 * 208, (long long)197 * 208,
         197, 197, 64, 8, 256, 0.125f, 0);
  softmax_plain_kernel<<<dim3(197, 256), 256, 0, stream>>>(
      fsc, 208, (long long)197 * 208, fpr, 208, (long long)197 * 208, 197);
  gemm16(fpr, 208, (long long)8 * 197 * 208, (long long)197 * 208,
         qkvh + 1024, 32 * 1536, 1536, 64,
         fO, 32 * 512, 512, 64, 197, 64, 197, 8, 256, 1.0f, 1);
  gemm(fO, 512, wh_aout, 512, a_outb, yb, 512, y2, nullptr, 512, YR, 512, 512, 0);

  // --- 10. final MLP ---
  ln(y2, ln2g, ln2b, ylb, YR);
  gemm(ylb, 512, wh_fc, 512, fcb, nullptr, 0, hb, nullptr, 2048, YR, 2048, 512, 1);
  gemm(hb, 2048, wh_pw, 2048, pb, y2, 512, y3, nullptr, 512, YR, 512, 2048, 0);

  // --- 11. scatter to (B,T,L,D) output ---
  scatter_out_kernel<<<TOK, 256, 0, stream>>>(y3, out);
}
